// HGSA_v14_3693671875060
// MI455X (gfx1250) — matrix operand order repaired and checked by reading, not compiled, not run on hardware
//
#include <hip/hip_runtime.h>
#include <math.h>

// ---------------- constants ----------------
#define BB     2
#define HH     256
#define WW     256
#define CC     64
#define NPIX   (HH*WW)           // 65536
#define MROWS  (BB*NPIX)         // 131072
#define HEADS  4
#define DHEAD  16

typedef _Float16 v16h __attribute__((ext_vector_type(16)));
typedef _Float16 v8h  __attribute__((ext_vector_type(8)));
typedef float    v8f  __attribute__((ext_vector_type(8)));
typedef unsigned int u32x4 __attribute__((ext_vector_type(4)));
typedef int          i32x8 __attribute__((ext_vector_type(8)));
typedef int          i32x4 __attribute__((ext_vector_type(4)));

// ---------------- WMMA fragment helpers (gfx1250, wave32) ----------------
// A (16x32 f16): lane m = lane&15, lh = lane>>4; halves j<8 -> k = k0+lh*8+j,
// j>=8 -> k = k0+16+lh*8+(j-8). Two contiguous 8-half (16B) LDS loads.
__device__ __forceinline__ v16h load_a_frag(const _Float16* base, int r0, int k0,
                                            int lm, int lh) {
    const _Float16* p = base + (r0 + lm) * 64 + k0 + lh * 8;
    union { v16h v; v8h h[2]; } u;
    u.h[0] = *(const v8h*)p;
    u.h[1] = *(const v8h*)(p + 16);
    return u.v;
}

// B (32x16 f16) from WT[k][64] (k-major): lane covers n = nt*16 + (lane&15);
// halves j<8 -> k = ks*32+lh*8+j, j>=8 -> k = ks*32+16+lh*8+(j-8) (strided gather).
__device__ __forceinline__ v16h load_b_frag(const _Float16* wt, int nt, int ks, int lane) {
    const _Float16* p = wt + (ks * 32 + (lane >> 4) * 8) * 64 + nt * 16 + (lane & 15);
    v16h v;
#pragma unroll
    for (int j = 0; j < 16; ++j) v[j] = p[((j & 7) + (j >> 3) * 16) * 64];
    return v;
}

__device__ __forceinline__ v8f wmma16(v16h a, v16h b, v8f c) {
    return __builtin_amdgcn_wmma_f32_16x16x32_f16(false, a, false, b, (short)0, c,
                                                  false, false);
}

// Tensor Data Mover: 1-D contiguous f16 tile, global -> LDS.
// D# built per CDNA5 ISA §8.3/8.4: group0 = {count=1, lds_addr, global_addr, type=2},
// group1 = {data_size=2B, tensor_dim0=tile_dim0=n, tile_dim1=1, stride0=n}.
// This toolchain exposes the 6-arg builtin (g0, g1, g2, g3, g4, cpol).
__device__ __forceinline__ void tdm_load_1d_f16(void* lds_dst, const void* src,
                                                unsigned int n_half) {
    unsigned long long ga = (unsigned long long)(uintptr_t)src;
    unsigned int lds_off  = (unsigned int)(uintptr_t)lds_dst;  // LDS byte offset
    u32x4 g0;
    g0[0] = 1u;                                                // count=1 (valid D#)
    g0[1] = lds_off;                                           // lds_addr [63:32]
    g0[2] = (unsigned int)(ga & 0xffffffffu);                  // global_addr lo
    g0[3] = (unsigned int)((ga >> 32) & 0x01ffffffu) | (2u << 30); // addr hi | type=2
    i32x8 g1;
    g1[0] = (int)(1u << 16);                          // workgroup_mask=0, data_size=1 (2B)
    g1[1] = (int)((n_half & 0xffffu) << 16);          // tensor_dim0[15:0] @ bits 63:48
    g1[2] = (int)((n_half >> 16) | (1u << 16));       // tensor_dim0[31:16] | tensor_dim1=1
    g1[3] = (int)((n_half & 0xffffu) << 16);          // tile_dim0 @ bits 127:112
    g1[4] = 1;                                        // tile_dim1=1, tile_dim2=0
    g1[5] = (int)n_half;                              // tensor_dim0_stride lo32
    g1[6] = (int)((n_half & 0xffffu) << 16);          // stride0 hi=0 | stride1[15:0]
    g1[7] = (int)(n_half >> 16);                      // stride1 hi
    i32x4 z4 = {0, 0, 0, 0};
    i32x8 z8 = {0, 0, 0, 0, 0, 0, 0, 0};
    __builtin_amdgcn_tensor_load_to_lds(g0, g1, z4, z4, z8, 0);
}

// ---------------- K0: zero accumulators ----------------
__global__ void hgsa_zero(float* accum) {
    for (int i = threadIdx.x; i < 2304; i += blockDim.x) accum[i] = 0.0f;
}

// ---------------- K1: fused QKV / m1 / m2 GEMMs + Gram & norm accumulation ----
// 128 rows per workgroup, 8 waves, 16-row M-tile per wave.
__global__ void hgsa_qkv(const float* __restrict__ x, const float* __restrict__ mask,
                         const float* __restrict__ Wq, const float* __restrict__ Wk,
                         const float* __restrict__ Wv, const float* __restrict__ mw1,
                         const float* __restrict__ mb1, const float* __restrict__ mw2,
                         const float* __restrict__ mb2,
                         _Float16* __restrict__ vout, _Float16* __restrict__ m1out,
                         _Float16* __restrict__ m2out, float* __restrict__ accum) {
    extern __shared__ char smem[];
    _Float16* Xl  = (_Float16*)smem;                    // 128*64 f16
    _Float16* Ml  = Xl + 128 * 64;                      // 128*64 f16
    _Float16* WTl = Ml + 128 * 64;                      // 5 * 64*64 f16 (k-major)
    float*    Ql  = (float*)(WTl + 5 * 64 * 64);        // 128*64 f32
    float*    Kl  = Ql + 128 * 64;                      // 128*64 f32
    _Float16* M1l = (_Float16*)(Kl + 128 * 64);         // 128*64 f16

    const int t    = threadIdx.x;
    const int m0   = blockIdx.x * 128;
    const int bidx = m0 >> 16;                          // batch index

    // stage activations (f32 -> f16)
    for (int i = t; i < 128 * 64; i += 256) {
        Xl[i] = (_Float16)x[(size_t)m0 * 64 + i];
        Ml[i] = (_Float16)mask[(size_t)m0 * 64 + i];
    }
    // stage transposed weights: WT[k*64+o] = W[o*64+k]
    for (int i = t; i < 64 * 64; i += 256) {
        int k = i >> 6, o = i & 63;
        WTl[0 * 4096 + i] = (_Float16)Wq[o * 64 + k];
        WTl[1 * 4096 + i] = (_Float16)Wk[o * 64 + k];
        WTl[2 * 4096 + i] = (_Float16)Wv[o * 64 + k];
        WTl[3 * 4096 + i] = (_Float16)mw1[o * 64 + k];
        WTl[4 * 4096 + i] = (_Float16)mw2[o * 64 + k];
    }
    __syncthreads();

    const int wave = t >> 5, lane = t & 31;
    const int r0 = wave * 16;
    const int lm = lane & 15, lh = lane >> 4;

    v16h ax0 = load_a_frag(Xl, r0, 0, lm, lh);
    v16h ax1 = load_a_frag(Xl, r0, 32, lm, lh);
    v16h am0 = load_a_frag(Ml, r0, 0, lm, lh);
    v16h am1 = load_a_frag(Ml, r0, 32, lm, lh);

#pragma unroll
    for (int nt = 0; nt < 4; ++nt) {
        // q
        v8f cq = {};
        cq = wmma16(ax0, load_b_frag(WTl + 0 * 4096, nt, 0, lane), cq);
        cq = wmma16(ax1, load_b_frag(WTl + 0 * 4096, nt, 1, lane), cq);
        // k
        v8f ck = {};
        ck = wmma16(ax0, load_b_frag(WTl + 1 * 4096, nt, 0, lane), ck);
        ck = wmma16(ax1, load_b_frag(WTl + 1 * 4096, nt, 1, lane), ck);
        // v
        v8f cv = {};
        cv = wmma16(ax0, load_b_frag(WTl + 2 * 4096, nt, 0, lane), cv);
        cv = wmma16(ax1, load_b_frag(WTl + 2 * 4096, nt, 1, lane), cv);
        // m1 = mask @ mw1^T + mb1
        v8f c1 = {};
        c1 = wmma16(am0, load_b_frag(WTl + 3 * 4096, nt, 0, lane), c1);
        c1 = wmma16(am1, load_b_frag(WTl + 3 * 4096, nt, 1, lane), c1);
        float b1v = mb1[nt * 16 + lm];
#pragma unroll
        for (int r = 0; r < 8; ++r) {
            int row = r0 + r + 8 * lh;
            int col = nt * 16 + lm;
            Ql[row * 64 + col] = cq[r];
            Kl[row * 64 + col] = ck[r];
            vout[(size_t)(m0 + row) * 64 + col] = (_Float16)cv[r];
            float m1v = c1[r] + b1v;
            M1l[row * 64 + col] = (_Float16)m1v;
            m1out[(size_t)(m0 + row) * 64 + col] = (_Float16)m1v;
        }
    }
    __syncthreads();

    // m2 = m1 @ mw2^T + mb2
    v16h a10 = load_a_frag(M1l, r0, 0, lm, lh);
    v16h a11 = load_a_frag(M1l, r0, 32, lm, lh);
#pragma unroll
    for (int nt = 0; nt < 4; ++nt) {
        v8f c2 = {};
        c2 = wmma16(a10, load_b_frag(WTl + 4 * 4096, nt, 0, lane), c2);
        c2 = wmma16(a11, load_b_frag(WTl + 4 * 4096, nt, 1, lane), c2);
        float b2v = mb2[nt * 16 + lm];
#pragma unroll
        for (int r = 0; r < 8; ++r) {
            int row = r0 + r + 8 * lh;
            m2out[(size_t)(m0 + row) * 64 + nt * 16 + lm] = (_Float16)(c2[r] + b2v);
        }
    }

    // Gram: S[b,h,d,e] += sum_p Kl[p][h*16+d] * Ql[p][h*16+e]
    {
        const int head = t >> 6;
        const int d    = (t >> 2) & 15;
        const int e0   = (t & 3) * 4;
        float acc0 = 0.f, acc1 = 0.f, acc2 = 0.f, acc3 = 0.f;
        for (int p = 0; p < 128; ++p) {
            float kv = Kl[p * 64 + head * 16 + d];
            const float* qp = &Ql[p * 64 + head * 16 + e0];
            acc0 = fmaf(kv, qp[0], acc0);
            acc1 = fmaf(kv, qp[1], acc1);
            acc2 = fmaf(kv, qp[2], acc2);
            acc3 = fmaf(kv, qp[3], acc3);
        }
        float* gram = accum + ((bidx * 4 + head) * 16 + d) * 16 + e0;
        atomicAdd(&gram[0], acc0);
        atomicAdd(&gram[1], acc1);
        atomicAdd(&gram[2], acc2);
        atomicAdd(&gram[3], acc3);
    }
    // sum of squares for L2 norms (q: threads 0..63, k: threads 64..127)
    if (t < 128) {
        int isK = t >> 6;
        int hh2 = (t >> 4) & 3;
        int e   = t & 15;
        const float* src = isK ? Kl : Ql;
        float s = 0.f;
        for (int p = 0; p < 128; ++p) {
            float vv = src[p * 64 + hh2 * 16 + e];
            s = fmaf(vv, vv, s);
        }
        atomicAdd(&accum[2048 + isK * 128 + (bidx * 4 + hh2) * 16 + e], s);
    }
}

// ---------------- K2: normalize + softmax + fold attn into Wp -> Weff^T -----
__global__ void hgsa_attn(const float* __restrict__ accum,
                          const float* __restrict__ rescale,
                          const float* __restrict__ Wp,
                          _Float16* __restrict__ wefft) {
    __shared__ float attnL[BB * HEADS * 16 * 16];
    const int t = threadIdx.x;
    if (t < 128) {                    // one row (b,h,d) of the 16x16 attn tile
        int b = t >> 6, head = (t >> 4) & 3, d = t & 15;
        const float* S   = accum + ((b * 4 + head) * 16 + d) * 16;
        const float* ssq = accum + 2048 + (b * 4 + head) * 16;
        const float* ssk = accum + 2176 + (b * 4 + head) * 16;
        float kn = fmaxf(sqrtf(ssk[d]), 1e-12f);
        float rs = rescale[head];
        float vals[16];
        float mx = -3.4e38f;
#pragma unroll
        for (int e = 0; e < 16; ++e) {
            float qn = fmaxf(sqrtf(ssq[e]), 1e-12f);
            vals[e] = S[e] / (kn * qn) * rs;
            mx = fmaxf(mx, vals[e]);
        }
        float sum = 0.f;
#pragma unroll
        for (int e = 0; e < 16; ++e) { vals[e] = expf(vals[e] - mx); sum += vals[e]; }
        float inv = 1.0f / sum;
#pragma unroll
        for (int e = 0; e < 16; ++e)
            attnL[(((b * 4 + head) * 16 + d) * 16) + e] = vals[e] * inv;
    }
    __syncthreads();
    // Weff^T[b][he][o] = sum_d Wp[o][h*16+d] * attn[b][h][d][e], he = h*16+e
    for (int idx = t; idx < BB * 64 * 64; idx += blockDim.x) {
        int b  = idx >> 12;
        int he = (idx >> 6) & 63;
        int o  = idx & 63;
        int h  = he >> 4, e = he & 15;
        float s = 0.f;
#pragma unroll
        for (int d = 0; d < 16; ++d)
            s = fmaf(Wp[o * 64 + h * 16 + d],
                     attnL[((b * 4 + h) * 16 + d) * 16 + e], s);
        wefft[idx] = (_Float16)s;
    }
}

// ---------------- K3: mask-gate stencil (5x5 dw + sigmoid) on v, PE path ----
// 16x16 pixel tile per block, 256 threads (one pixel each).
__global__ void hgsa_stencil(const float* __restrict__ x,
                             const _Float16* __restrict__ m1g,
                             const _Float16* __restrict__ m2g,
                             _Float16* __restrict__ vg,
                             const float* __restrict__ mdw,
                             const float* __restrict__ mdwb,
                             const float* __restrict__ pw1,
                             const float* __restrict__ pw2,
                             float* __restrict__ out) {
    extern __shared__ char smem[];
    _Float16* m2s  = (_Float16*)smem;            // 20*20*64 f16
    _Float16* t2s  = m2s + 20 * 20 * 64;         // 18*18*64 f16
    float*    wdw  = (float*)(t2s + 18 * 18 * 64); // 64*25
    float*    wp1  = wdw + 64 * 25;              // 64*9
    float*    wp2  = wp1 + 64 * 9;               // 64*9
    float*    wdwb = wp2 + 64 * 9;               // 64

    const int t  = threadIdx.x;
    const int bi = blockIdx.x;
    const int b  = bi >> 8;
    const int by = (bi >> 4) & 15;
    const int bx = bi & 15;

    for (int i = t; i < 64 * 25; i += 256) wdw[i] = mdw[i];
    for (int i = t; i < 64 * 9; i += 256) { wp1[i] = pw1[i]; wp2[i] = pw2[i]; }
    if (t < 64) wdwb[t] = mdwb[t];

    // stage m2 with halo 2 (zero padded)
    for (int i = t; i < 20 * 20 * 64; i += 256) {
        int iy = i / (20 * 64);
        int r  = i - iy * (20 * 64);
        int ix = r >> 6, c = r & 63;
        int gy = by * 16 - 2 + iy, gx = bx * 16 - 2 + ix;
        _Float16 v = (_Float16)0.0f;
        if (gy >= 0 && gy < HH && gx >= 0 && gx < WW)
            v = m2g[((size_t)(b * HH + gy) * WW + gx) * 64 + c];
        m2s[i] = v;
    }
    // compute t2 = gelu(dwconv3(x)) with halo 1
    for (int i = t; i < 18 * 18 * 64; i += 256) {
        int iy = i / (18 * 64);
        int r  = i - iy * (18 * 64);
        int ix = r >> 6, c = r & 63;
        int gy = by * 16 - 1 + iy, gx = bx * 16 - 1 + ix;
        float t1 = 0.f;
#pragma unroll
        for (int ky = 0; ky < 3; ++ky) {
#pragma unroll
            for (int kx = 0; kx < 3; ++kx) {
                int yy = gy + ky - 1, xx = gx + kx - 1;
                if (yy >= 0 && yy < HH && xx >= 0 && xx < WW)
                    t1 = fmaf(x[((size_t)(b * HH + yy) * WW + xx) * 64 + c],
                              wp1[c * 9 + ky * 3 + kx], t1);
            }
        }
        float g = 0.5f * t1 * (1.0f + erff(t1 * 0.7071067811865476f));
        t2s[i] = (_Float16)g;
    }
    __syncthreads();

    const int ly = t >> 4, lx = t & 15;
    const int y = by * 16 + ly, xq = bx * 16 + lx;
    const size_t pix = (size_t)(b * HH + y) * WW + xq;
    __builtin_prefetch(&m1g[pix * 64], 0, 1);
    __builtin_prefetch(&vg[pix * 64], 0, 1);

    for (int c = 0; c < 64; ++c) {
        // depthwise 5x5 on m2, sigmoid
        float s = wdwb[c];
#pragma unroll
        for (int ky = 0; ky < 5; ++ky) {
#pragma unroll
            for (int kx = 0; kx < 5; ++kx)
                s = fmaf((float)m2s[((ly + ky) * 20 + (lx + kx)) * 64 + c],
                         wdw[c * 25 + ky * 5 + kx], s);
        }
        float sig = 1.0f / (1.0f + expf(-s));
        float m1v = (float)m1g[pix * 64 + c];
        float ma  = fmaf(m1v, sig, m1v);
        float vf  = (float)vg[pix * 64 + c] * ma;
        vg[pix * 64 + c] = (_Float16)vf;     // in-place mask-gated v

        // depthwise 3x3 on gelu output
        float p = 0.f;
#pragma unroll
        for (int ky = 0; ky < 3; ++ky) {
#pragma unroll
            for (int kx = 0; kx < 3; ++kx)
                p = fmaf((float)t2s[((ly + ky) * 18 + (lx + kx)) * 64 + c],
                         wp2[c * 9 + ky * 3 + kx], p);
        }
        out[pix * 64 + c] = p;               // positional encoding; K4 adds GEMM
    }
}

// ---------------- K4: out = pe + v @ Weff^T + bp (WMMA, TDM-staged) ---------
__global__ void hgsa_proj(const _Float16* __restrict__ vg,
                          const _Float16* __restrict__ wefft,
                          const float* __restrict__ bp,
                          float* __restrict__ out) {
    __shared__ _Float16 Vl[128 * 64];
    __shared__ _Float16 Wl[64 * 64];
    const int t  = threadIdx.x;
    const int m0 = blockIdx.x * 128;
    const int b  = m0 >> 16;

    // Tensor Data Mover: DMA both contiguous tiles into LDS (one wave issues;
    // TENSORcnt tracks completion, barrier publishes to the workgroup).
    if (t < 32) {
        tdm_load_1d_f16(Vl, vg + (size_t)m0 * 64, 128 * 64);
        tdm_load_1d_f16(Wl, wefft + b * 4096, 64 * 64);
        __builtin_amdgcn_s_wait_tensorcnt(0);
    }
    asm volatile("" ::: "memory");
    __syncthreads();

    const int wave = t >> 5, lane = t & 31;
    const int r0 = wave * 16;
    const int lm = lane & 15, lh = lane >> 4;

    v16h a0 = load_a_frag(Vl, r0, 0, lm, lh);
    v16h a1 = load_a_frag(Vl, r0, 32, lm, lh);
#pragma unroll
    for (int nt = 0; nt < 4; ++nt) {
        v8f c = {};
        c = wmma16(a0, load_b_frag(Wl, nt, 0, lane), c);
        c = wmma16(a1, load_b_frag(Wl, nt, 1, lane), c);
        float bv = bp[nt * 16 + lm];
#pragma unroll
        for (int r = 0; r < 8; ++r) {
            size_t idx = (size_t)(m0 + r0 + r + 8 * lh) * 64 + nt * 16 + lm;
            out[idx] = out[idx] + c[r] + bv;
        }
    }
}

// ---------------- launcher ----------------
extern "C" void kernel_launch(void* const* d_in, const int* in_sizes, int n_in,
                              void* d_out, int out_size, void* d_ws, size_t ws_size,
                              hipStream_t stream) {
    const float* x    = (const float*)d_in[0];
    const float* mask = (const float*)d_in[1];
    const float* Wq   = (const float*)d_in[2];
    const float* Wk   = (const float*)d_in[3];
    const float* Wv   = (const float*)d_in[4];
    const float* resc = (const float*)d_in[5];
    const float* Wp   = (const float*)d_in[6];
    const float* bp   = (const float*)d_in[7];
    const float* mw1  = (const float*)d_in[8];
    const float* mb1  = (const float*)d_in[9];
    const float* mw2  = (const float*)d_in[10];
    const float* mb2  = (const float*)d_in[11];
    const float* mdw  = (const float*)d_in[12];
    const float* mdwb = (const float*)d_in[13];
    const float* pw1  = (const float*)d_in[14];
    const float* pw2  = (const float*)d_in[15];

    char* ws = (char*)d_ws;
    _Float16* vbuf  = (_Float16*)(ws);                       // 16 MB
    _Float16* m1buf = (_Float16*)(ws + (size_t)16777216);    // 16 MB
    _Float16* m2buf = (_Float16*)(ws + (size_t)33554432);    // 16 MB
    float*    accum = (float*)(ws + (size_t)50331648);       // 2304 f32
    _Float16* wefft = (_Float16*)(ws + (size_t)50340864);    // 8192 f16
    float* out = (float*)d_out;

    hgsa_zero<<<1, 256, 0, stream>>>(accum);

    size_t lds1 = (size_t)(128 * 64 * 2) * 2 + 5 * 64 * 64 * 2 +
                  (size_t)(128 * 64 * 4) * 2 + 128 * 64 * 2;   // 155648 B
    hgsa_qkv<<<MROWS / 128, 256, lds1, stream>>>(x, mask, Wq, Wk, Wv, mw1, mb1,
                                                 mw2, mb2, vbuf, m1buf, m2buf,
                                                 accum);

    hgsa_attn<<<1, 256, 0, stream>>>(accum, resc, Wp, wefft);

    size_t lds3 = (size_t)(20 * 20 * 64 * 2) + 18 * 18 * 64 * 2 +
                  64 * 25 * 4 + 64 * 9 * 4 * 2 + 64 * 4;       // 103936 B
    hgsa_stencil<<<BB * 16 * 16, 256, lds3, stream>>>(x, m1buf, m2buf, vbuf,
                                                      mdw, mdwb, pw1, pw2, out);

    hgsa_proj<<<MROWS / 128, 256, 0, stream>>>(vbuf, wefft, bp, out);
}
